// PNSMotorNet_17738214932679
// MI455X (gfx1250) — compile-verified, matrix operand order checked by reading, not run on hardware
//
#include <hip/hip_runtime.h>

// ---------------------------------------------------------------------------
// Routed MoE linear: y[i] = action[i] @ W[e].T + b[e],  e = expert_idx[i]
// B=16384, E=16, D=512, all f32.
//   1) sort samples by expert (histogram + scatter permutation)
//   2) pre-split W into bf16 hi/lo planes in workspace (once per call)
//   3) per 16-row expert tile: bf16-split WMMA GEMM (hi*hi + hi*lo + lo*hi),
//      f32 accumulation on v_wmma_f32_16x16x32_bf16; 2 N-tiles per wave.
// ---------------------------------------------------------------------------

typedef __attribute__((ext_vector_type(16))) __bf16 v16bf;
typedef __attribute__((ext_vector_type(8)))  float  v8f;

#define B_N 16384
#define E_N 16
#define D_N 512
#define MAXT (B_N / 16 + E_N)   // 1040 worst-case row tiles

// Workspace layout:
//  int32 region:
//   [0..15]                 counts
//   [16..32]                offsets (17)
//   [33..48]                cursor
//   [49]                    totalTiles
//   [64 .. 64+MAXT)         tileExpert
//   [64+MAXT .. 64+2*MAXT)  tileRowStart
//   [4096 .. 4096+B_N)      perm
//  uint32 planes (pairs of bf16 packed along d):
//   [WS_WHI_U ..)           Whi  E*D*D/2 uints (8 MB)
//   [WS_WLO_U ..)           Wlo  E*D*D/2 uints (8 MB)
#define WS_COUNTS   0
#define WS_OFFSETS  16
#define WS_CURSOR   33
#define WS_TOTAL    49
#define WS_TILE_E   64
#define WS_TILE_R   (64 + MAXT)
#define WS_PERM     4096
#define WS_WHI_U    32768                       // 128 KB in
#define WS_PLANE_U  (E_N * D_N * D_N / 2)       // 2,097,152 uints
#define WS_WLO_U    (WS_WHI_U + WS_PLANE_U)
#define WS_NEEDED_BYTES ((size_t)(WS_WHI_U + 2 * WS_PLANE_U) * 4)

// --- bf16 split helper ------------------------------------------------------
__device__ __forceinline__ void split2(float a, float b,
                                       unsigned int& hi, unsigned int& lo) {
    __bf16 ha = (__bf16)a, hb = (__bf16)b;
    __bf16 la = (__bf16)(a - (float)ha), lb = (__bf16)(b - (float)hb);
    union { unsigned int u; __bf16 h[2]; } ph, pl;
    ph.h[0] = ha; ph.h[1] = hb;
    pl.h[0] = la; pl.h[1] = lb;
    hi = ph.u; lo = pl.u;
}

union FragU { v16bf v; unsigned int u[8]; uint4 q[2]; };

// --- pass 1: zero counters --------------------------------------------------
__global__ void moe_init_kernel(int* wsI) {
    int t = threadIdx.x;
    if (t < E_N) { wsI[WS_COUNTS + t] = 0; wsI[WS_CURSOR + t] = 0; }
    if (t == E_N) wsI[WS_TOTAL] = 0;
}

// --- pass 2: histogram ------------------------------------------------------
__global__ void moe_hist_kernel(const int* __restrict__ expert_idx, int* wsI) {
    int i = blockIdx.x * blockDim.x + threadIdx.x;
    if (i < B_N) atomicAdd(&wsI[WS_COUNTS + expert_idx[i]], 1);
}

// --- pass 3: serial plan (16 experts, <=1040 tiles) -------------------------
__global__ void moe_plan_kernel(int* wsI) {
    if (threadIdx.x != 0 || blockIdx.x != 0) return;
    int off = 0;
    for (int e = 0; e < E_N; ++e) {
        wsI[WS_OFFSETS + e] = off;
        wsI[WS_CURSOR + e]  = off;
        off += wsI[WS_COUNTS + e];
    }
    wsI[WS_OFFSETS + E_N] = off;
    int t = 0;
    for (int e = 0; e < E_N; ++e) {
        int base = wsI[WS_OFFSETS + e];
        int cnt  = wsI[WS_COUNTS + e];
        for (int j = 0; j < cnt; j += 16) {
            wsI[WS_TILE_E + t] = e;
            wsI[WS_TILE_R + t] = base + j;
            ++t;
        }
    }
    wsI[WS_TOTAL] = t;
}

// --- pass 4: scatter into expert-grouped permutation ------------------------
__global__ void moe_scatter_kernel(const int* __restrict__ expert_idx, int* wsI) {
    int i = blockIdx.x * blockDim.x + threadIdx.x;
    if (i < B_N) {
        int pos = atomicAdd(&wsI[WS_CURSOR + expert_idx[i]], 1);
        wsI[WS_PERM + pos] = i;
    }
}

// --- pass 5 (optional): pre-split W into bf16 hi/lo planes ------------------
// One thread handles 4 consecutive f32 along d -> 2 packed-pair uints/plane.
__global__ void moe_wsplit_kernel(const float* __restrict__ W, unsigned int* wsU) {
    size_t t = (size_t)blockIdx.x * blockDim.x + threadIdx.x;  // float4 index
    const float4 wv = *(const float4*)(W + t * 4);
    unsigned int h0, l0, h1, l1;
    split2(wv.x, wv.y, h0, l0);
    split2(wv.z, wv.w, h1, l1);
    uint2* hp = (uint2*)(wsU + WS_WHI_U);
    uint2* lp = (uint2*)(wsU + WS_WLO_U);
    hp[t] = make_uint2(h0, h1);
    lp[t] = make_uint2(l0, l1);
}

// --- pass 6: expert-tile GEMM with bf16-split WMMA --------------------------
// grid = (MAXT, D/128), block = 128 (4 waves).
// Wave w owns two 16x16 tiles at cols blockIdx.y*128 + w*32 (+0, +16).
// K-map (ISA 7.12.2, 16-bit frags): u[0..3] = pairs at (k0/2 + 4*half),
// u[4..7] = pairs at (k0/2 + 8 + 4*half)  -> two contiguous b128 loads.
template <bool PRESPLIT>
__global__ __launch_bounds__(128)
void moe_gemm_kernel(const float* __restrict__ X,           // [B, D]
                     const float* __restrict__ W,           // [E, D, D] (f32 path)
                     const unsigned int* __restrict__ Whi,  // packed bf16 pairs
                     const unsigned int* __restrict__ Wlo,
                     const float* __restrict__ bias,        // [E, D]
                     const int*   __restrict__ wsI,
                     float* __restrict__ out) {             // [B, D]
    const int totalTiles = wsI[WS_TOTAL];
    const int tile = blockIdx.x;
    if (tile >= totalTiles) return;                         // wave-uniform exit

    const int e        = wsI[WS_TILE_E + tile];
    const int rowStart = wsI[WS_TILE_R + tile];
    int valid = wsI[WS_OFFSETS + e + 1] - rowStart;
    if (valid > 16) valid = 16;

    const int tid  = threadIdx.x;
    const int wave = tid >> 5;
    const int lane = tid & 31;
    const int n    = lane & 15;    // N col (B/C/D) == M row (A)
    const int half = lane >> 4;

    __shared__ unsigned int lA_hi[16][20];  // bf16 pairs along k, 80B rows
    __shared__ unsigned int lA_lo[16][20];
    __shared__ int rows[16];

    if (tid < 16) rows[tid] = (tid < valid) ? wsI[WS_PERM + rowStart + tid] : -1;
    __syncthreads();

    const int col0 = blockIdx.y * 128 + wave * 32;          // this wave's 2 tiles
    const int c0   = col0 + n;                              // tile 0 column
    const int c1   = col0 + 16 + n;                         // tile 1 column

    // Per-column fragment base pointers (uint pair index = (e*D + col)*D/2)
    const size_t pb0 = ((size_t)e * D_N + c0) * (D_N / 2);
    const size_t pb1 = ((size_t)e * D_N + c1) * (D_N / 2);
    const float* Wp0 = W + ((size_t)e * D_N + c0) * D_N;    // f32 fallback
    const float* Wp1 = W + ((size_t)e * D_N + c1) * D_N;

    // A staging: thread t loads 4 consecutive f32 of one gathered row
    const int ar = tid >> 3;          // 0..15
    const int ak = (tid & 7) * 4;     // 0..28
    const float* Xp = (rows[ar] >= 0) ? (X + (size_t)rows[ar] * D_N) : nullptr;

    v8f acc0 = {}, acc1 = {};

    for (int k0 = 0; k0 < D_N; k0 += 32) {
        // ---- stage A tile: 16x32 f32 -> bf16 hi/lo pairs in LDS ----
        float x0 = 0.f, x1 = 0.f, x2 = 0.f, x3 = 0.f;
        if (Xp) {
            const float4 xv = *(const float4*)(Xp + k0 + ak);
            x0 = xv.x; x1 = xv.y; x2 = xv.z; x3 = xv.w;
        }
        unsigned int h0, l0, h1, l1;
        split2(x0, x1, h0, l0);
        split2(x2, x3, h1, l1);
        const int p = ak >> 1;                              // even -> 8B aligned
        *(uint2*)&lA_hi[ar][p] = make_uint2(h0, h1);
        *(uint2*)&lA_lo[ar][p] = make_uint2(l0, l1);
        __syncthreads();

        // ---- A fragments: two b128 per plane ----
        FragU a_hi, a_lo;
        a_hi.q[0] = *(const uint4*)&lA_hi[n][4 * half];
        a_hi.q[1] = *(const uint4*)&lA_hi[n][8 + 4 * half];
        a_lo.q[0] = *(const uint4*)&lA_lo[n][4 * half];
        a_lo.q[1] = *(const uint4*)&lA_lo[n][8 + 4 * half];

        // ---- B fragments for both tiles ----
        FragU b0_hi, b0_lo, b1_hi, b1_lo;
        if (PRESPLIT) {
            const size_t kp = (size_t)(k0 >> 1) + 4 * half;
            b0_hi.q[0] = *(const uint4*)(Whi + pb0 + kp);
            b0_hi.q[1] = *(const uint4*)(Whi + pb0 + kp + 8);
            b0_lo.q[0] = *(const uint4*)(Wlo + pb0 + kp);
            b0_lo.q[1] = *(const uint4*)(Wlo + pb0 + kp + 8);
            b1_hi.q[0] = *(const uint4*)(Whi + pb1 + kp);
            b1_hi.q[1] = *(const uint4*)(Whi + pb1 + kp + 8);
            b1_lo.q[0] = *(const uint4*)(Wlo + pb1 + kp);
            b1_lo.q[1] = *(const uint4*)(Wlo + pb1 + kp + 8);
        } else {
#pragma unroll
            for (int v = 0; v < 8; ++v) {
                const int kk = (v < 4) ? (2 * v + 8 * half)
                                       : (16 + 2 * (v - 4) + 8 * half);
                const float2 w0 = *(const float2*)(Wp0 + k0 + kk);
                const float2 w1 = *(const float2*)(Wp1 + k0 + kk);
                split2(w0.x, w0.y, b0_hi.u[v], b0_lo.u[v]);
                split2(w1.x, w1.y, b1_hi.u[v], b1_lo.u[v]);
            }
        }

        // ---- 3-term split products, f32 accumulate ----
        acc0 = __builtin_amdgcn_wmma_f32_16x16x32_bf16(
                   false, a_hi.v, false, b0_hi.v, (short)0, acc0, false, false);
        acc1 = __builtin_amdgcn_wmma_f32_16x16x32_bf16(
                   false, a_hi.v, false, b1_hi.v, (short)0, acc1, false, false);
        acc0 = __builtin_amdgcn_wmma_f32_16x16x32_bf16(
                   false, a_hi.v, false, b0_lo.v, (short)0, acc0, false, false);
        acc1 = __builtin_amdgcn_wmma_f32_16x16x32_bf16(
                   false, a_hi.v, false, b1_lo.v, (short)0, acc1, false, false);
        acc0 = __builtin_amdgcn_wmma_f32_16x16x32_bf16(
                   false, a_lo.v, false, b0_hi.v, (short)0, acc0, false, false);
        acc1 = __builtin_amdgcn_wmma_f32_16x16x32_bf16(
                   false, a_lo.v, false, b1_hi.v, (short)0, acc1, false, false);

        __syncthreads();
    }

    // ---- epilogue: +bias, scatter rows through the permutation ----
    const float bb0 = bias[e * D_N + c0];
    const float bb1 = bias[e * D_N + c1];
#pragma unroll
    for (int v = 0; v < 8; ++v) {
        const int m = v + 8 * half;          // C/D layout: VGPRv holds M=v / v+8
        if (m < valid) {
            float* orow = out + (size_t)rows[m] * D_N;
            orow[c0] = acc0[v] + bb0;
            orow[c1] = acc1[v] + bb1;
        }
    }
}

extern "C" void kernel_launch(void* const* d_in, const int* in_sizes, int n_in,
                              void* d_out, int out_size, void* d_ws, size_t ws_size,
                              hipStream_t stream) {
    const float* action     = (const float*)d_in[0];   // [B, D]
    const float* W          = (const float*)d_in[1];   // [E, D, D]
    const float* bias       = (const float*)d_in[2];   // [E, D]
    const int*   expert_idx = (const int*)d_in[3];     // [B]
    float*       out        = (float*)d_out;           // [B, D]
    int*          wsI       = (int*)d_ws;
    unsigned int* wsU       = (unsigned int*)d_ws;

    moe_init_kernel<<<1, 64, 0, stream>>>(wsI);
    moe_hist_kernel<<<B_N / 256, 256, 0, stream>>>(expert_idx, wsI);
    moe_plan_kernel<<<1, 1, 0, stream>>>(wsI);
    moe_scatter_kernel<<<B_N / 256, 256, 0, stream>>>(expert_idx, wsI);

    const bool presplit = (ws_size >= WS_NEEDED_BYTES);   // fixed per run
    dim3 grid(MAXT, D_N / 128, 1);                        // 1040 x 4
    if (presplit) {
        moe_wsplit_kernel<<<E_N * D_N * D_N / (256 * 4), 256, 0, stream>>>(W, wsU);
        moe_gemm_kernel<true><<<grid, 128, 0, stream>>>(
            action, W, wsU + WS_WHI_U, wsU + WS_WLO_U, bias, wsI, out);
    } else {
        moe_gemm_kernel<false><<<grid, 128, 0, stream>>>(
            action, W, wsU + WS_WHI_U, wsU + WS_WLO_U, bias, wsI, out);
    }
}
